// GNNQNet_17617955848486
// MI455X (gfx1250) — compile-verified, matrix operand order checked by reading, not run on hardware
//
#include <hip/hip_runtime.h>
#include <hip/hip_bf16.h>
#include <math.h>

// ---------------------------------------------------------------------------
// Problem constants (from reference)
// ---------------------------------------------------------------------------
#define NN    4096
#define EE    131072
#define INDIM 256
#define HID   128
#define NHEAD 4
#define DD    512          // HID*NHEAD
#define EDIM  3
#define NCOL  32
#define BN_EPS 1e-5f

typedef __attribute__((ext_vector_type(16))) __bf16 v16bf;
typedef __attribute__((ext_vector_type(8)))  float  v8f;

// ---------------------------------------------------------------------------
// Small helpers
// ---------------------------------------------------------------------------
__device__ __forceinline__ void atomicMaxF(float* addr, float val) {
    unsigned int* a = (unsigned int*)addr;
    unsigned int old = __float_as_uint(*addr);
    while (__uint_as_float(old) < val) {
        unsigned int assumed = old;
        old = atomicCAS(a, assumed, __float_as_uint(val));
        if (old == assumed) break;
    }
}

__global__ void fill_kernel(float* p, float v, long n) {
    long i = (long)blockIdx.x * blockDim.x + threadIdx.x;
    if (i < n) p[i] = v;
}

__global__ void cvt_kernel(const float* __restrict__ in, __bf16* __restrict__ out, long n) {
    long i = (long)blockIdx.x * blockDim.x + threadIdx.x;
    if (i < n) out[i] = (__bf16)in[i];
}

// transpose-convert: in[R][C] f32  ->  out[C][R] bf16
__global__ void cvt_t_kernel(const float* __restrict__ in, __bf16* __restrict__ out,
                             int R, int C) {
    long i = (long)blockIdx.x * blockDim.x + threadIdx.x;
    if (i >= (long)R * C) return;
    int r = (int)(i / C), c = (int)(i % C);
    out[(size_t)c * R + r] = (__bf16)in[i];
}

// ---------------------------------------------------------------------------
// GEMM: C[M,Nc] = A[M,K](bf16,row-major) x Bt(bf16, pre-transposed: Bt[n][k]),
// f32 accumulate via v_wmma_f32_16x16x32_bf16.
// Tile 64x128, BK=64, 256 threads (8 waves), wave = 32x32 patch, 8 WMMA/round.
// All tile traffic is b128 vectors (no scalar LDS transposes).
// Fused epilogue: +bias, +residual, ReLU, f32 / bf16 / V-transposed stores.
// ---------------------------------------------------------------------------
#define LSB 80   // LDS row stride in halves (160B: 32B-aligned, bank-skewed)

__global__ __launch_bounds__(256)
void gemm_bf16_kernel(const __bf16* __restrict__ A, const __bf16* __restrict__ Bt,
                      int M, int K, int Nc,
                      const float* __restrict__ bias,
                      const float* __restrict__ res,
                      int do_relu,
                      float* __restrict__ outF, __bf16* __restrict__ outB,
                      __bf16* __restrict__ outVT)   // transposed store of cols >= 2*DD
{
    __shared__ __align__(32) __bf16 As[64][LSB];
    __shared__ __align__(32) __bf16 Bs[128][LSB];

    const int tid  = threadIdx.x;
    const int wave = tid >> 5;
    const int lane = tid & 31;
    const int hl   = lane >> 4;      // K-half selector within a 32-wide k step
    const int ln   = lane & 15;      // row/col within fragment
    const int m0   = blockIdx.y * 64;
    const int n0   = blockIdx.x * 128;
    const int wm   = (wave >> 2) * 32;   // 0 / 32
    const int wn   = (wave & 3) * 32;    // 0,32,64,96

    // cooperative-load coordinates
    const int ar = tid >> 2;             // 0..63   A row
    const int ac = (tid & 3) << 4;       // 0..48   A col (16 halves)
    const int bn = tid >> 1;             // 0..127  B row (n)
    const int bc = (tid & 1) << 5;       // 0 / 32  B col (32 halves)

    v8f acc[2][2] = {};

    for (int kt = 0; kt < K; kt += 64) {
        // ---- cooperative vector loads (global b128 -> LDS b128) ----
        {
            const __bf16* ap = A + (size_t)(m0 + ar) * K + kt + ac;
            *(uint4*)&As[ar][ac]     = *(const uint4*)(ap);
            *(uint4*)&As[ar][ac + 8] = *(const uint4*)(ap + 8);
            if (kt + 64 < K) __builtin_prefetch(ap + 64, 0, 0);   // global_prefetch
        }
        {
            const __bf16* bp = Bt + (size_t)(n0 + bn) * K + kt + bc;
            *(uint4*)&Bs[bn][bc]      = *(const uint4*)(bp);
            *(uint4*)&Bs[bn][bc + 8]  = *(const uint4*)(bp + 8);
            *(uint4*)&Bs[bn][bc + 16] = *(const uint4*)(bp + 16);
            *(uint4*)&Bs[bn][bc + 24] = *(const uint4*)(bp + 24);
            if (kt + 64 < K) __builtin_prefetch(bp + 64, 0, 0);
        }
        __syncthreads();

        #pragma unroll
        for (int ks = 0; ks < 64; ks += 32) {
            // fragments: lane = row, hl = K-half (contiguous k = 16*hl + s, same order in A and B)
            v16bf af0 = *(const v16bf*)&As[wm +      ln][ks + hl * 16];
            v16bf af1 = *(const v16bf*)&As[wm + 16 + ln][ks + hl * 16];
            v16bf bf0 = *(const v16bf*)&Bs[wn +      ln][ks + hl * 16];
            v16bf bf1 = *(const v16bf*)&Bs[wn + 16 + ln][ks + hl * 16];

            acc[0][0] = __builtin_amdgcn_wmma_f32_16x16x32_bf16(false, af0, false, bf0,
                                                                (short)0, acc[0][0], false, false);
            acc[0][1] = __builtin_amdgcn_wmma_f32_16x16x32_bf16(false, af0, false, bf1,
                                                                (short)0, acc[0][1], false, false);
            acc[1][0] = __builtin_amdgcn_wmma_f32_16x16x32_bf16(false, af1, false, bf0,
                                                                (short)0, acc[1][0], false, false);
            acc[1][1] = __builtin_amdgcn_wmma_f32_16x16x32_bf16(false, af1, false, bf1,
                                                                (short)0, acc[1][1], false, false);
        }
        __syncthreads();
    }

    // ---- epilogue (C/D layout: lane ln = column, rows r + 8*hl) ----
    #pragma unroll
    for (int i = 0; i < 2; ++i)
        #pragma unroll
        for (int j = 0; j < 2; ++j)
            #pragma unroll
            for (int r = 0; r < 8; ++r) {
                int row = m0 + wm + i * 16 + r + 8 * hl;
                int col = n0 + wn + j * 16 + ln;
                float v = acc[i][j][r];
                if (bias)  v += bias[col];
                if (res)   v += res[(size_t)row * Nc + col];
                if (do_relu) v = fmaxf(v, 0.f);
                size_t idx = (size_t)row * Nc + col;
                if (outF) outF[idx] = v;
                if (outB) outB[idx] = (__bf16)v;
                if (outVT && col >= 2 * DD)     // V slice, stored [h*dh][node]
                    outVT[(size_t)(col - 2 * DD) * M + row] = (__bf16)v;
            }
}

// ---------------------------------------------------------------------------
// Flash attention over all nodes (seq = 4096, 4 heads, dh = 128).
// qkvb: [N][3*D] bf16 (q | k | v rows); vtb: [D][N] bf16 (V pre-transposed).
// Grid (N/64, NHEAD), 128 threads. Each wave owns 16 query rows and streams
// K/V in 64-row chunks through LDS; scores never touch HBM.
// ---------------------------------------------------------------------------
__global__ __launch_bounds__(128)
void mha_flash_kernel(const __bf16* __restrict__ qkvb, const __bf16* __restrict__ vtb,
                      float* __restrict__ ctx)
{
    __shared__ __align__(32) __bf16 Ks[64][144];   // K chunk, row-major  (stride 288B)
    __shared__ __align__(32) __bf16 Vt[128][80];   // V chunk, transposed (stride 160B)
    __shared__ __align__(32) __bf16 Pw[4][16][80]; // per-wave P re-layout strip

    const int tid  = threadIdx.x;
    const int wave = tid >> 5;
    const int lane = tid & 31;
    const int hl   = lane >> 4;
    const int ln   = lane & 15;
    const int head = blockIdx.y;
    const int rowBase = blockIdx.x * 64 + wave * 16;
    const float scale = 0.08838834764831845f;      // 1/sqrt(128)

    // Q fragments for this wave's 16 rows (16x128 = 4 chunks of K=32)
    v16bf qf[4];
    #pragma unroll
    for (int c = 0; c < 4; ++c)
        qf[c] = *(const v16bf*)(qkvb + (size_t)(rowBase + ln) * (3 * DD)
                                + head * HID + c * 32 + hl * 16);

    float mrow[8], lrow[8];
    v8f acc[8] = {};
    #pragma unroll
    for (int r = 0; r < 8; ++r) { mrow[r] = -1e30f; lrow[r] = 0.f; }

    for (int j0 = 0; j0 < NN; j0 += 64) {
        // ---- cooperative vector loads: K row-major, V already transposed ----
        #pragma unroll
        for (int i = 0; i < 8; ++i) {               // K: 64x128 halves
            int chunk = tid + i * 128;              // 0..1023
            int r  = chunk >> 4;                    // 0..63
            int c8 = (chunk & 15) << 3;             // 0..120
            const __bf16* kp = qkvb + (size_t)(j0 + r) * (3 * DD) + DD + head * HID + c8;
            *(uint4*)&Ks[r][c8] = *(const uint4*)kp;
        }
        #pragma unroll
        for (int i = 0; i < 8; ++i) {               // Vt: 128x64 halves
            int chunk = tid + i * 128;              // 0..1023
            int r  = chunk >> 3;                    // 0..127 (dh)
            int c8 = (chunk & 7) << 3;              // 0..56  (node)
            const __bf16* vp = vtb + (size_t)(head * HID + r) * NN + j0 + c8;
            *(uint4*)&Vt[r][c8] = *(const uint4*)vp;
        }
        __syncthreads();

        // ---- S = Q K^T  (16x64 tile, 16 WMMA) ----
        v8f s[4] = {};
        #pragma unroll
        for (int nf = 0; nf < 4; ++nf)
            #pragma unroll
            for (int c = 0; c < 4; ++c) {
                v16bf kb = *(const v16bf*)&Ks[nf * 16 + ln][c * 32 + hl * 16];
                s[nf] = __builtin_amdgcn_wmma_f32_16x16x32_bf16(false, qf[c], false, kb,
                                                                (short)0, s[nf], false, false);
            }

        // ---- online softmax (reductions across 16-lane halves) ----
        float mnew[8], alf[8];
        #pragma unroll
        for (int r = 0; r < 8; ++r) {
            float t = -1e30f;
            #pragma unroll
            for (int nf = 0; nf < 4; ++nf) t = fmaxf(t, s[nf][r] * scale);
            #pragma unroll
            for (int off = 8; off >= 1; off >>= 1)
                t = fmaxf(t, __shfl_xor(t, off, 32));
            mnew[r] = fmaxf(mrow[r], t);
            alf[r]  = __expf(mrow[r] - mnew[r]);
            mrow[r] = mnew[r];
        }
        #pragma unroll
        for (int nf = 0; nf < 4; ++nf)
            #pragma unroll
            for (int r = 0; r < 8; ++r) {
                float p = __expf(s[nf][r] * scale - mnew[r]);
                s[nf][r] = p;
                Pw[wave][r + 8 * hl][nf * 16 + ln] = (__bf16)p;
            }
        #pragma unroll
        for (int r = 0; r < 8; ++r) {
            float t = s[0][r] + s[1][r] + s[2][r] + s[3][r];
            #pragma unroll
            for (int off = 8; off >= 1; off >>= 1) t += __shfl_xor(t, off, 32);
            lrow[r] = lrow[r] * alf[r] + t;
        }
        #pragma unroll
        for (int d = 0; d < 8; ++d)
            #pragma unroll
            for (int r = 0; r < 8; ++r) acc[d][r] *= alf[r];

        // ---- O += P V  (16 WMMA) ----
        v16bf pa0 = *(const v16bf*)&Pw[wave][ln][hl * 16];
        v16bf pa1 = *(const v16bf*)&Pw[wave][ln][32 + hl * 16];
        #pragma unroll
        for (int d = 0; d < 8; ++d) {
            v16bf vb0 = *(const v16bf*)&Vt[d * 16 + ln][hl * 16];
            v16bf vb1 = *(const v16bf*)&Vt[d * 16 + ln][32 + hl * 16];
            acc[d] = __builtin_amdgcn_wmma_f32_16x16x32_bf16(false, pa0, false, vb0,
                                                             (short)0, acc[d], false, false);
            acc[d] = __builtin_amdgcn_wmma_f32_16x16x32_bf16(false, pa1, false, vb1,
                                                             (short)0, acc[d], false, false);
        }
        __syncthreads();
    }

    #pragma unroll
    for (int d = 0; d < 8; ++d)
        #pragma unroll
        for (int r = 0; r < 8; ++r) {
            int row = rowBase + r + 8 * hl;
            ctx[(size_t)row * DD + head * HID + d * 16 + ln] = acc[d][r] / lrow[r];
        }
}

// ---------------------------------------------------------------------------
// GAT auxiliary kernels (memory-bound; f32 atomics, trivial vs 23.3 TB/s)
// ---------------------------------------------------------------------------
__global__ void edge_deg_kernel(const int* __restrict__ dst, const float* __restrict__ eattr,
                                float* __restrict__ cnt, float* __restrict__ attrsum) {
    int e = blockIdx.x * blockDim.x + threadIdx.x;
    if (e >= EE) return;
    int d = dst[e];
    atomicAdd(&cnt[d], 1.f);
    atomicAdd(&attrsum[d * 3 + 0], eattr[e * 3 + 0]);
    atomicAdd(&attrsum[d * 3 + 1], eattr[e * 3 + 1]);
    atomicAdd(&attrsum[d * 3 + 2], eattr[e * 3 + 2]);
}

__global__ void loopattr_kernel(const float* __restrict__ cnt, const float* __restrict__ attrsum,
                                float* __restrict__ loopattr) {
    int i = blockIdx.x * blockDim.x + threadIdx.x;
    if (i >= NN * 3) return;
    int n = i / 3;
    loopattr[i] = attrsum[i] / fmaxf(cnt[n], 1.f);
}

// collapse edge attention: Medge[k*4+h] = sum_c W_edge[k][h*128+c] * att_edge[h][c]
__global__ void medge_kernel(const float* __restrict__ W_edge, const float* __restrict__ att_edge,
                             float* __restrict__ Mout) {
    int t = threadIdx.x;
    if (t >= EDIM * NHEAD) return;
    int k = t / NHEAD, h = t % NHEAD;
    float s = 0.f;
    for (int c = 0; c < HID; ++c)
        s += W_edge[k * DD + h * HID + c] * att_edge[h * HID + c];
    Mout[t] = s;
}

__global__ void node_att_kernel(const float* __restrict__ h,
                                const float* __restrict__ att_src, const float* __restrict__ att_dst,
                                float* __restrict__ asrc, float* __restrict__ adst) {
    int i = blockIdx.x * blockDim.x + threadIdx.x;     // N*NHEAD
    if (i >= NN * NHEAD) return;
    int n = i >> 2, hd = i & 3;
    const float* hp = h + (size_t)n * DD + hd * HID;
    const float* as = att_src + hd * HID;
    const float* ad = att_dst + hd * HID;
    float s1 = 0.f, s2 = 0.f;
    for (int c = 0; c < HID; ++c) { float v = hp[c]; s1 += v * as[c]; s2 += v * ad[c]; }
    asrc[i] = s1; adst[i] = s2;
}

__global__ void edge_alpha1_kernel(const int* __restrict__ src, const int* __restrict__ dst,
                                   const float* __restrict__ eattr, const float* __restrict__ loopattr,
                                   const float* __restrict__ Medge,
                                   const float* __restrict__ asrc, const float* __restrict__ adst,
                                   float* __restrict__ alphabuf, float* __restrict__ amax) {
    int i = blockIdx.x * blockDim.x + threadIdx.x;
    if (i >= EE + NN) return;
    int s, d; float e0, e1, e2;
    if (i < EE) { s = src[i]; d = dst[i];
        e0 = eattr[i * 3]; e1 = eattr[i * 3 + 1]; e2 = eattr[i * 3 + 2];
    } else { s = d = i - EE;
        e0 = loopattr[s * 3]; e1 = loopattr[s * 3 + 1]; e2 = loopattr[s * 3 + 2];
    }
    #pragma unroll
    for (int h = 0; h < NHEAD; ++h) {
        float a = asrc[s * 4 + h] + adst[d * 4 + h]
                + e0 * Medge[h] + e1 * Medge[4 + h] + e2 * Medge[8 + h];
        a = (a > 0.f) ? a : 0.2f * a;                  // leaky_relu(0.2)
        alphabuf[(size_t)i * 4 + h] = a;
        atomicMaxF(&amax[d * 4 + h], a);
    }
}

__global__ void edge_alpha2_kernel(const int* __restrict__ dst,
                                   float* __restrict__ alphabuf,
                                   const float* __restrict__ amax, float* __restrict__ denom) {
    int i = blockIdx.x * blockDim.x + threadIdx.x;
    if (i >= EE + NN) return;
    int d = (i < EE) ? dst[i] : i - EE;
    #pragma unroll
    for (int h = 0; h < NHEAD; ++h) {
        float e = __expf(alphabuf[(size_t)i * 4 + h] - amax[d * 4 + h]);
        alphabuf[(size_t)i * 4 + h] = e;
        atomicAdd(&denom[d * 4 + h], e);
    }
}

__global__ void edge_scatter_kernel(const int* __restrict__ src, const int* __restrict__ dst,
                                    const float* __restrict__ h, const float* __restrict__ alphabuf,
                                    const float* __restrict__ denom, float* __restrict__ agg) {
    long gid = (long)blockIdx.x * blockDim.x + threadIdx.x;
    if (gid >= (long)(EE + NN) * DD) return;
    int i = (int)(gid >> 9);
    int c = (int)(gid & (DD - 1));
    int hd = c >> 7;
    int s, d;
    if (i < EE) { s = src[i]; d = dst[i]; } else { s = d = i - EE; }
    float w = alphabuf[(size_t)i * 4 + hd] / (denom[d * 4 + hd] + 1e-16f);
    atomicAdd(&agg[(size_t)d * DD + c], h[(size_t)s * DD + c] * w);
}

__global__ void finalize_kernel(const float* __restrict__ agg, const float* __restrict__ bias,
                                const float* __restrict__ gamma, const float* __restrict__ beta,
                                const float* __restrict__ mean,  const float* __restrict__ var,
                                float* __restrict__ xf, __bf16* __restrict__ xb) {
    long i = (long)blockIdx.x * blockDim.x + threadIdx.x;
    if (i >= (long)NN * DD) return;
    int c = (int)(i & (DD - 1));
    float v = agg[i] + bias[c];
    v = (v - mean[c]) * gamma[c] * rsqrtf(var[c] + BN_EPS) + beta[c];
    v = fmaxf(v, 0.f);
    xf[i] = v;
    xb[i] = (__bf16)v;
}

// ---------------------------------------------------------------------------
// Pool (mean + max over nodes) and final MLP (tiny, M=1 -> VALU)
// ---------------------------------------------------------------------------
__global__ void pool_kernel(const float* __restrict__ x, float* __restrict__ g) {
    int c = blockIdx.x * blockDim.x + threadIdx.x;
    if (c >= DD) return;
    float s = 0.f, mx = -1e30f;
    for (int n = 0; n < NN; ++n) {
        float v = x[(size_t)n * DD + c];
        s += v; mx = fmaxf(mx, v);
    }
    g[c] = s / (float)NN;
    g[DD + c] = mx;
}

__global__ __launch_bounds__(256)
void mlp_kernel(const float* __restrict__ g,
                const float* __restrict__ W1, const float* __restrict__ b1,
                const float* __restrict__ W2, const float* __restrict__ b2,
                const float* __restrict__ W3, const float* __restrict__ b3,
                float* __restrict__ out) {
    __shared__ float gs[2 * DD];
    __shared__ float h1[2 * HID];
    __shared__ float h2[HID];
    int t = threadIdx.x;
    for (int i = t; i < 2 * DD; i += 256) gs[i] = g[i];
    __syncthreads();
    {   // h1 = relu(g @ W1 + b1)   W1: [1024][256]
        float s = b1[t];
        for (int k = 0; k < 2 * DD; ++k) s += gs[k] * W1[k * (2 * HID) + t];
        h1[t] = fmaxf(s, 0.f);
    }
    __syncthreads();
    if (t < HID) {   // h2 = relu(h1 @ W2 + b2)   W2: [256][128]
        float s = b2[t];
        for (int k = 0; k < 2 * HID; ++k) s += h1[k] * W2[k * HID + t];
        h2[t] = fmaxf(s, 0.f);
    }
    __syncthreads();
    if (t < 3 * NCOL) {   // q = h2 @ W3 + b3   W3: [128][96]
        float s = b3[t];
        for (int k = 0; k < HID; ++k) s += h2[k] * W3[k * (3 * NCOL) + t];
        out[t] = s;
    }
}

// ---------------------------------------------------------------------------
// Host-side orchestration
// ---------------------------------------------------------------------------
static inline int gblk(long n, int b) { return (int)((n + b - 1) / b); }

extern "C" void kernel_launch(void* const* d_in, const int* in_sizes, int n_in,
                              void* d_out, int out_size, void* d_ws, size_t ws_size,
                              hipStream_t stream) {
    (void)in_sizes; (void)n_in; (void)out_size; (void)ws_size;

    // ---- input pointers (setup_inputs dict order, nested insertion order) ----
    const float* x_in  = (const float*)d_in[0];
    const int*   ei    = (const int*)d_in[1];
    const float* eattr = (const float*)d_in[2];
    const int* srcI = ei;
    const int* dstI = ei + EE;
    // gat layer l: 3 + l*6 -> {W, W_edge, att_src, att_dst, att_edge, bias}
    // bn  layer l: 21 + l*4 -> {gamma, beta, mean, var}
    const float* W_in  = (const float*)d_in[33];
    const float* b_in  = (const float*)d_in[34];
    const float* W_out = (const float*)d_in[35];
    const float* b_out = (const float*)d_in[36];
    const float* W1 = (const float*)d_in[37]; const float* b1 = (const float*)d_in[38];
    const float* W2 = (const float*)d_in[39]; const float* b2 = (const float*)d_in[40];
    const float* W3 = (const float*)d_in[41]; const float* b3 = (const float*)d_in[42];

    // ---- workspace carve ----
    char* wp = (char*)d_ws;
    auto carve = [&](size_t bytes) -> void* {
        void* p = (void*)wp;
        wp += (bytes + 255) & ~(size_t)255;
        return p;
    };
    float*  xbuf     = (float*)carve((size_t)NN * DD * 4);       // activations f32
    float*  hbuf     = (float*)carve((size_t)NN * DD * 4);       // h = x@W ; reused as ctx
    float*  aggbuf   = (float*)carve((size_t)NN * DD * 4);       // segment-sum / final x
    float*  alphabuf = (float*)carve((size_t)(EE + NN) * NHEAD * 4);
    float*  asrc     = (float*)carve((size_t)NN * NHEAD * 4);
    float*  adst     = (float*)carve((size_t)NN * NHEAD * 4);
    float*  amax     = (float*)carve((size_t)NN * NHEAD * 4);
    float*  denom    = (float*)carve((size_t)NN * NHEAD * 4);
    float*  cnt      = (float*)carve((size_t)NN * 4);
    float*  attrsum  = (float*)carve((size_t)NN * 3 * 4);
    float*  loopattr = (float*)carve((size_t)NN * 3 * 4);
    float*  Medge    = (float*)carve(64);
    float*  gpool    = (float*)carve(2 * DD * 4);
    __bf16* xb       = (__bf16*)carve((size_t)NN * DD * 2);      // bf16 activations
    __bf16* wb       = (__bf16*)carve((size_t)DD * 3 * DD * 2);  // bf16 weights (Bt layout)
    __bf16* qkvb     = (__bf16*)carve((size_t)NN * 3 * DD * 2);  // bf16 qkv
    __bf16* vtb      = (__bf16*)carve((size_t)NN * DD * 2);      // bf16 V transposed [D][N]

    // ---- self-loop edge attr (layer-independent) ----
    fill_kernel<<<gblk(NN, 256), 256, 0, stream>>>(cnt, 0.f, NN);
    fill_kernel<<<gblk((long)NN * 3, 256), 256, 0, stream>>>(attrsum, 0.f, (long)NN * 3);
    edge_deg_kernel<<<gblk(EE, 256), 256, 0, stream>>>(dstI, eattr, cnt, attrsum);
    loopattr_kernel<<<gblk((long)NN * 3, 256), 256, 0, stream>>>(cnt, attrsum, loopattr);

    // ---- x -> bf16 ----
    cvt_kernel<<<gblk((long)NN * INDIM, 256), 256, 0, stream>>>(x_in, xb, (long)NN * INDIM);

    int inDim = INDIM;
    for (int l = 0; l < 3; ++l) {
        const float* W     = (const float*)d_in[3 + l * 6 + 0];
        const float* Wedge = (const float*)d_in[3 + l * 6 + 1];
        const float* attS  = (const float*)d_in[3 + l * 6 + 2];
        const float* attD  = (const float*)d_in[3 + l * 6 + 3];
        const float* attE  = (const float*)d_in[3 + l * 6 + 4];
        const float* biasL = (const float*)d_in[3 + l * 6 + 5];
        const float* gam   = (const float*)d_in[21 + l * 4 + 0];
        const float* bet   = (const float*)d_in[21 + l * 4 + 1];
        const float* mea   = (const float*)d_in[21 + l * 4 + 2];
        const float* va    = (const float*)d_in[21 + l * 4 + 3];

        // h = x @ W : W[inDim][D] -> Bt[D][inDim] once, then WMMA GEMM
        cvt_t_kernel<<<gblk((long)inDim * DD, 256), 256, 0, stream>>>(W, wb, inDim, DD);
        gemm_bf16_kernel<<<dim3(DD / 128, NN / 64), 256, 0, stream>>>(
            xb, wb, NN, inDim, DD,
            nullptr, nullptr, 0, hbuf, nullptr, nullptr);

        node_att_kernel<<<gblk((long)NN * NHEAD, 256), 256, 0, stream>>>(hbuf, attS, attD, asrc, adst);
        medge_kernel<<<1, 64, 0, stream>>>(Wedge, attE, Medge);

        fill_kernel<<<gblk((long)NN * NHEAD, 256), 256, 0, stream>>>(amax, -1e30f, (long)NN * NHEAD);
        fill_kernel<<<gblk((long)NN * NHEAD, 256), 256, 0, stream>>>(denom, 0.f, (long)NN * NHEAD);
        fill_kernel<<<gblk((long)NN * DD, 256), 256, 0, stream>>>(aggbuf, 0.f, (long)NN * DD);

        edge_alpha1_kernel<<<gblk(EE + NN, 256), 256, 0, stream>>>(
            srcI, dstI, eattr, loopattr, Medge, asrc, adst, alphabuf, amax);
        edge_alpha2_kernel<<<gblk(EE + NN, 256), 256, 0, stream>>>(dstI, alphabuf, amax, denom);
        edge_scatter_kernel<<<gblk((long)(EE + NN) * DD, 256), 256, 0, stream>>>(
            srcI, dstI, hbuf, alphabuf, denom, aggbuf);

        finalize_kernel<<<gblk((long)NN * DD, 256), 256, 0, stream>>>(
            aggbuf, biasL, gam, bet, mea, va, xbuf, xb);
        inDim = DD;
    }

    // ---- MHA: qkv = x @ W_in^T + b_in.
    // W_in is [3D][D] row-major == Bt[n][k] layout directly (no transpose needed).
    cvt_kernel<<<gblk((long)3 * DD * DD, 256), 256, 0, stream>>>(W_in, wb, (long)3 * DD * DD);
    gemm_bf16_kernel<<<dim3(3 * DD / 128, NN / 64), 256, 0, stream>>>(
        xb, wb, NN, DD, 3 * DD,
        b_in, nullptr, 0, nullptr, qkvb, vtb);      // also emit V transposed

    // flash attention -> ctx (reuse hbuf)
    mha_flash_kernel<<<dim3(NN / 64, NHEAD), 128, 0, stream>>>(qkvb, vtb, hbuf);

    // out proj + residual: x = x + ctx @ W_out^T + b_out.
    // W_out [D][D] row-major == Bt layout directly.
    cvt_kernel<<<gblk((long)NN * DD, 256), 256, 0, stream>>>(hbuf, xb, (long)NN * DD);
    cvt_kernel<<<gblk((long)DD * DD, 256), 256, 0, stream>>>(W_out, wb, (long)DD * DD);
    gemm_bf16_kernel<<<dim3(DD / 128, NN / 64), 256, 0, stream>>>(
        xb, wb, NN, DD, DD,
        b_out, xbuf, 0, aggbuf, nullptr, nullptr);

    // ---- pool + MLP head ----
    pool_kernel<<<gblk(DD, 256), 256, 0, stream>>>(aggbuf, gpool);
    mlp_kernel<<<1, 256, 0, stream>>>(gpool, W1, b1, W2, b2, W3, b3, (float*)d_out);
}
